// LocalAttentionDecoder_22840636080334
// MI455X (gfx1250) — matrix operand order repaired and checked by reading, not compiled, not run on hardware
//
#include <hip/hip_runtime.h>
#include <hip/hip_bf16.h>
#include <math.h>

// Problem constants
#define NB   16
#define LSRC 64
#define HID  1024
#define VOC  32000
#define TDEC 63
#define DEVP 50.0f

typedef __attribute__((ext_vector_type(16))) __bf16 v16bf;
typedef __attribute__((ext_vector_type(8)))  float  v8f;
typedef __attribute__((ext_vector_type(4)))  int    v4i;

#if __has_builtin(__builtin_amdgcn_global_load_async_to_lds_b128) && \
    __has_builtin(__builtin_amdgcn_s_wait_asynccnt)
#define HAVE_ASYNC_LDS 1
#else
#define HAVE_ASYNC_LDS 0
#endif

// Copy 16B global -> LDS. Async (ASYNCcnt-tracked) when the toolchain exposes it.
__device__ __forceinline__ void cp16_to_lds(unsigned short* dst_lds, const unsigned short* src_glb) {
#if HAVE_ASYNC_LDS
  __builtin_amdgcn_global_load_async_to_lds_b128(
      (__attribute__((address_space(1))) v4i*)(unsigned short*)src_glb,
      (__attribute__((address_space(3))) v4i*)dst_lds,
      0, 0);
#else
  *(v4i*)dst_lds = *(const v4i*)src_glb;
#endif
}

__device__ __forceinline__ void lds_copy_wait() {
#if HAVE_ASYNC_LDS
  __builtin_amdgcn_s_wait_asynccnt(0);
#endif
  __syncthreads();
}

// ---------------- fragment layout helpers (CDNA5 WMMA 16x16x32 bf16) ----------------
// A matrix 16x32 (MxK), 16 bf16 per lane (ISA 7.12.2):
//   lanes 0-15 : M=lane, K=0..7 in slots 0..7,  K=16..23 in slots 8..15
//   lanes 16-31: M=lane-16, K=8..15 in slots 0..7, K=24..31 in slots 8..15
__device__ __forceinline__ size_t apack(int n, int k) {
  int kt = k >> 5, kk = k & 31;
  int lane = n + (((kk >> 3) & 1) << 4);
  int slot = (kk & 7) + ((kk & 16) >> 1);
  return (((size_t)kt * 32 + lane) << 4) + slot;
}
// B matrix 32x16 (KxN): row K striped across lanes (N=lane%16);
//   lanes 0-15 hold K=0..7,16..23, lanes 16-31 hold K=8..15,24..31 (same slots as A).
// Packed N-tile-major, K-tile-minor so the K loop streams contiguously.
__device__ __forceinline__ size_t bpack(int k, int m, int Ktiles) {
  int kt = k >> 5, kk = k & 31, nt = m >> 4;
  int lane = (m & 15) + (((kk >> 3) & 1) << 4);
  int slot = (kk & 7) + ((kk & 16) >> 1);
  return ((((size_t)nt * Ktiles + kt) * 32 + lane) << 4) + slot;
}

__device__ __forceinline__ unsigned short f2bf(float f) {
  unsigned int u = __float_as_uint(f);
  unsigned int r = u + 0x7FFFu + ((u >> 16) & 1u);   // round-nearest-even
  return (unsigned short)(r >> 16);
}

__device__ __forceinline__ v16bf ldf(const unsigned short* p) {
  return *(const v16bf*)p;
}

__device__ __forceinline__ float sigm(float x) { return 1.0f / (1.0f + __expf(-x)); }

#define WMMA_BF16(A, B, C) \
  __builtin_amdgcn_wmma_f32_16x16x32_bf16(false, (A), false, (B), (short)0, (C), false, false)

// ---------------- kernel 0: fp32 -> bf16 weight packing ----------------
__global__ void __launch_bounds__(256)
k_pack(const float* Wih0, const float* Whh0, const float* Wih1, const float* Whh1,
       const float* Wa, const float* Wc, const float* Wo,
       unsigned short* W0p, unsigned short* W1p, unsigned short* Wap,
       unsigned short* Wcp, unsigned short* Wop) {
  const size_t n0 = (size_t)3072 * 4096;   // [W_ih0 | W_hh0]^T, K=3072, M=4096
  const size_t n1 = (size_t)2048 * 4096;   // [W_ih1 | W_hh1]^T, K=2048, M=4096
  const size_t na = (size_t)1024 * 1024;   // W_a^T
  const size_t nc = (size_t)2048 * 1024;   // W_c^T
  const size_t no = (size_t)1024 * 32000;  // W_out^T
  const size_t total = n0 + n1 + na + nc + no;
  size_t stride = (size_t)gridDim.x * blockDim.x;
  for (size_t e = (size_t)blockIdx.x * blockDim.x + threadIdx.x; e < total; e += stride) {
    if (e < n0) {
      int k = (int)(e % 3072); int m = (int)(e / 3072);
      float v = (k < 2048) ? Wih0[(size_t)m * 2048 + k] : Whh0[(size_t)m * 1024 + (k - 2048)];
      W0p[bpack(k, m, 96)] = f2bf(v);
    } else if (e < n0 + n1) {
      size_t i = e - n0; int k = (int)(i % 2048); int m = (int)(i / 2048);
      float v = (k < 1024) ? Wih1[(size_t)m * 1024 + k] : Whh1[(size_t)m * 1024 + (k - 1024)];
      W1p[bpack(k, m, 64)] = f2bf(v);
    } else if (e < n0 + n1 + na) {
      size_t i = e - n0 - n1; int k = (int)(i & 1023); int m = (int)(i >> 10);
      Wap[bpack(k, m, 32)] = f2bf(Wa[(size_t)m * 1024 + k]);
    } else if (e < n0 + n1 + na + nc) {
      size_t i = e - n0 - n1 - na; int k = (int)(i & 2047); int m = (int)(i >> 11);
      Wcp[bpack(k, m, 64)] = f2bf(Wc[(size_t)m * 2048 + k]);
    } else {
      size_t i = e - n0 - n1 - na - nc; int k = (int)(i & 1023); int m = (int)(i >> 10);
      Wop[bpack(k, m, 32)] = f2bf(Wo[(size_t)m * 1024 + k]);
    }
  }
}

// ---------------- init: carry state + first-step A fragments ----------------
__global__ void __launch_bounds__(256)
k_init(const int* target, const float* embedding, const float* h_0, const float* c_0,
       unsigned short* XF, unsigned short* H1F0,
       float* h0, float* c0, float* h1, float* c1) {
  int stride = gridDim.x * blockDim.x;
  int tid = blockIdx.x * blockDim.x + threadIdx.x;
  for (int e = tid; e < NB * HID; e += stride) {
    h0[e] = h_0[e];
    h1[e] = h_0[NB * HID + e];
    c0[e] = c_0[e];
    c1[e] = c_0[NB * HID + e];
    int n = e >> 10, k = e & 1023;
    H1F0[apack(n, k)] = f2bf(h_0[NB * HID + e]);
  }
  for (int e = tid; e < NB * 3072; e += stride) {
    int n = e / 3072, k = e % 3072;
    float v;
    if (k < 1024) { int tok = target[n * (TDEC + 1)]; v = embedding[(size_t)tok * HID + k]; }
    else if (k < 2048) v = 0.0f;                       // input-feeding attn starts at zero
    else v = h_0[(size_t)n * HID + (k - 2048)];        // h0 carry
    XF[apack(n, k)] = f2bf(v);
  }
}

// ---------------- LSTM layer 0: gates = [x|h0] @ [Wih0|Whh0]^T + b ----------------
// A fragments are staged into LDS in 32-Ktile (32KB) chunks shared by all 8 waves.
__global__ void __launch_bounds__(256)
k_lstm0(const unsigned short* XF, const unsigned short* W0p,
        const float* bih, const float* bhh,
        float* h0, float* c0, unsigned short* H0F) {
  __shared__ unsigned short sA[32 * 32 * 16];    // 32 ktiles x 32 lanes x 16 bf16
  int tid = threadIdx.x;
  int lane = tid & 31;
  int wave = tid >> 5;
  int hNt = blockIdx.x * 8 + wave;               // 0..63 : one h-column tile per wave
  int col = (hNt << 4) + (lane & 15);
  v8f acc[4];
#pragma unroll
  for (int g = 0; g < 4; ++g) {
    float b = bih[g * HID + col] + bhh[g * HID + col];
#pragma unroll
    for (int j = 0; j < 8; ++j) acc[g][j] = b;
  }
  for (int chunk = 0; chunk < 3; ++chunk) {      // 96 ktiles in 3 chunks of 32
    for (int i = tid; i < 2048; i += 256)        // 2048 x 16B = 32KB
      cp16_to_lds(&sA[i * 8], XF + ((size_t)chunk * 16384) + (size_t)i * 8);
    lds_copy_wait();
    for (int kt = 0; kt < 32; ++kt) {
      v16bf a = ldf(&sA[((kt * 32 + lane) << 4)]);
      int ktg = chunk * 32 + kt;
#pragma unroll
      for (int g = 0; g < 4; ++g) {
        int nt = g * 64 + hNt;                   // i,f,g,o gate blocks of 1024
        v16bf b = ldf(W0p + ((((size_t)nt * 96 + ktg) * 32 + lane) << 4));
        acc[g] = WMMA_BF16(a, b, acc[g]);
      }
    }
    __syncthreads();
  }
#pragma unroll
  for (int j = 0; j < 8; ++j) {
    int n = j + ((lane >> 4) << 3);              // D layout: rows j / j+8
    size_t si = (size_t)n * HID + col;
    float cold = c0[si];
    float iv = sigm(acc[0][j]), fv = sigm(acc[1][j]);
    float gv = tanhf(acc[2][j]), ov = sigm(acc[3][j]);
    float cn = fv * cold + iv * gv;
    float hn = ov * tanhf(cn);
    c0[si] = cn; h0[si] = hn;
    H0F[apack(n, col)] = f2bf(hn);
  }
}

// ---------------- LSTM layer 1 ----------------
__global__ void __launch_bounds__(256)
k_lstm1(const unsigned short* H0F, const unsigned short* H1Fcur,
        const unsigned short* W1p, const float* bih, const float* bhh,
        float* h1, float* c1, unsigned short* H1Fnext) {
  __shared__ unsigned short sA[32 * 32 * 16];
  int tid = threadIdx.x;
  int lane = tid & 31;
  int wave = tid >> 5;
  int hNt = blockIdx.x * 8 + wave;
  int col = (hNt << 4) + (lane & 15);
  v8f acc[4];
#pragma unroll
  for (int g = 0; g < 4; ++g) {
    float b = bih[g * HID + col] + bhh[g * HID + col];
#pragma unroll
    for (int j = 0; j < 8; ++j) acc[g][j] = b;
  }
  for (int chunk = 0; chunk < 2; ++chunk) {      // chunk 0 = h0n frags, chunk 1 = h1 carry
    const unsigned short* src = (chunk == 0) ? H0F : H1Fcur;
    for (int i = tid; i < 2048; i += 256)
      cp16_to_lds(&sA[i * 8], src + (size_t)i * 8);
    lds_copy_wait();
    for (int kt = 0; kt < 32; ++kt) {
      v16bf a = ldf(&sA[((kt * 32 + lane) << 4)]);
      int ktg = chunk * 32 + kt;
#pragma unroll
      for (int g = 0; g < 4; ++g) {
        int nt = g * 64 + hNt;
        v16bf b = ldf(W1p + ((((size_t)nt * 64 + ktg) * 32 + lane) << 4));
        acc[g] = WMMA_BF16(a, b, acc[g]);
      }
    }
    __syncthreads();
  }
#pragma unroll
  for (int j = 0; j < 8; ++j) {
    int n = j + ((lane >> 4) << 3);
    size_t si = (size_t)n * HID + col;
    float cold = c1[si];
    float iv = sigm(acc[0][j]), fv = sigm(acc[1][j]);
    float gv = tanhf(acc[2][j]), ov = sigm(acc[3][j]);
    float cn = fv * cold + iv * gv;
    float hn = ov * tanhf(cn);
    c1[si] = cn; h1[si] = hn;
    H1Fnext[apack(n, col)] = f2bf(hn);           // double-buffered carry fragments
  }
}

// ---------------- attention + W_c + next-step fragment staging (1 block) ----------------
__global__ void __launch_bounds__(512)
k_attn(const unsigned short* Wap, const unsigned short* Wcp,
       const float* enc, const int* src_len, const int* target, const float* embedding,
       const float* h0, const float* h1, float* qbuf, float* ctxbuf,
       unsigned short* QF, unsigned short* CF, unsigned short* AF, unsigned short* XF,
       int t) {
  __shared__ float sA[NB][LSRC];
  int tid = threadIdx.x;
  int lane = tid & 31, wave = tid >> 5;

  // 1. stage q-GEMM A fragments from h1 (= LSTM output `out`)
  for (int e = tid; e < NB * HID; e += 512) {
    int n = e >> 10, k = e & 1023;
    QF[apack(n, k)] = f2bf(h1[e]);
  }
  __threadfence(); __syncthreads();

  // 2. q = out @ W_a^T  (M=1024, 16 waves x 4 tiles)
  for (int r = 0; r < 4; ++r) {
    int nt = wave * 4 + r;
    v8f acc = {};
    for (int kt = 0; kt < 32; ++kt) {
      v16bf a = ldf(QF + (((size_t)kt * 32 + lane) << 4));
      v16bf b = ldf(Wap + ((((size_t)nt * 32 + kt) * 32 + lane) << 4));
      acc = WMMA_BF16(a, b, acc);
    }
    int col = (nt << 4) + (lane & 15);
#pragma unroll
    for (int j = 0; j < 8; ++j) {
      int n = j + ((lane >> 4) << 3);
      qbuf[(size_t)n * HID + col] = acc[j];
    }
  }
  __threadfence(); __syncthreads();

  // 3. scores[n][l] = q[n] . enc[n][l]
  for (int p = tid; p < NB * LSRC; p += 512) {
    int n = p >> 6, l = p & 63;
    const float* qr = qbuf + (size_t)n * HID;
    const float* er = enc + ((size_t)(n * LSRC + l)) * HID;
    float s = 0.f;
    for (int h = 0; h < HID; ++h) s += qr[h] * er[h];
    sA[n][l] = s;
  }
  __syncthreads();

  // 4. softmax over L, then Gaussian local window (not renormalized)
  if (tid < NB) {
    int n = tid;
    float m = -1e30f;
    for (int l = 0; l < LSRC; ++l) m = fmaxf(m, sA[n][l]);
    float sum = 0.f;
    for (int l = 0; l < LSRC; ++l) { float e = __expf(sA[n][l] - m); sA[n][l] = e; sum += e; }
    float inv = 1.0f / sum;
    float pt = fmaxf((float)src_len[n] - (float)(t + 1), -1.0f);
    for (int l = 0; l < LSRC; ++l) {
      float d = (float)l - pt;
      sA[n][l] = sA[n][l] * inv * __expf(-(d * d) * (1.0f / DEVP));
    }
  }
  __syncthreads();

  // 5. ctx[n][h] = sum_l a[n][l] * enc[n][l][h]
  {
    int n = tid >> 5, hb = tid & 31;
    for (int c = 0; c < 32; ++c) {
      int h = hb + (c << 5);
      float s = 0.f;
      for (int l = 0; l < LSRC; ++l) s += sA[n][l] * enc[((size_t)(n * LSRC + l)) * HID + h];
      ctxbuf[(size_t)n * HID + h] = s;
    }
  }
  __threadfence(); __syncthreads();

  // 6. stage [ctx | out] fragments for W_c GEMM
  for (int e = tid; e < NB * 2048; e += 512) {
    int n = e >> 11, k = e & 2047;
    float v = (k < HID) ? ctxbuf[(size_t)n * HID + k] : h1[(size_t)n * HID + (k - HID)];
    CF[apack(n, k)] = f2bf(v);
  }
  __threadfence(); __syncthreads();

  // 7. attn = tanh([ctx|out] @ W_c^T); emit logits-A frags + next-step x attn part
  for (int r = 0; r < 4; ++r) {
    int nt = wave * 4 + r;
    v8f acc = {};
    for (int kt = 0; kt < 64; ++kt) {
      v16bf a = ldf(CF + (((size_t)kt * 32 + lane) << 4));
      v16bf b = ldf(Wcp + ((((size_t)nt * 64 + kt) * 32 + lane) << 4));
      acc = WMMA_BF16(a, b, acc);
    }
    int col = (nt << 4) + (lane & 15);
#pragma unroll
    for (int j = 0; j < 8; ++j) {
      int n = j + ((lane >> 4) << 3);
      unsigned short bv = f2bf(tanhf(acc[j]));
      AF[apack(n, col)] = bv;           // A-fragments for the W_out GEMM
      XF[apack(n, HID + col)] = bv;     // input-feeding slice of next step's x
    }
  }

  // 8. next-step x fragments: embedding of token t+1, and h0 carry
  if (t + 1 < TDEC) {
    __syncthreads();
    for (int e = tid; e < NB * 2048; e += 512) {
      int n = e >> 11, k = e & 2047;
      float v;
      int kx;
      if (k < HID) { int tok = target[n * (TDEC + 1) + (t + 1)]; v = embedding[(size_t)tok * HID + k]; kx = k; }
      else { v = h0[(size_t)n * HID + (k - HID)]; kx = k + HID; }
      XF[apack(n, kx)] = f2bf(v);
    }
  }
}

// ---------------- logits = attn @ W_out^T  (dominant GEMM, 250 blocks) ----------------
__global__ void __launch_bounds__(256)
k_logits(const unsigned short* AF, const unsigned short* Wop, float* out, int t) {
  __shared__ unsigned short sAF[32 * 32 * 16];   // full A-fragment set, 32KB
  int tid = threadIdx.x;
  int lane = tid & 31, wave = tid >> 5;
  for (int i = tid; i < 2048; i += 256)          // one async 32KB stage per block
    cp16_to_lds(&sAF[i * 8], AF + (size_t)i * 8);
  lds_copy_wait();

  int nt = blockIdx.x * 8 + wave;                // 0..1999 vocab tiles of 16
  v8f acc = {};
  const unsigned short* Bb = Wop + (((size_t)nt * 32 * 32 + lane) << 4);
  for (int kt = 0; kt < 32; ++kt) {
    v16bf a = ldf(&sAF[((kt * 32 + lane) << 4)]);
    v16bf b = ldf(Bb + ((size_t)kt << 9));
    __builtin_prefetch(Bb + (((size_t)kt + 2) << 9), 0, 0);   // global_prefetch_b8
    acc = WMMA_BF16(a, b, acc);
  }
  int v = (nt << 4) + (lane & 15);
#pragma unroll
  for (int j = 0; j < 8; ++j) {
    int n = j + ((lane >> 4) << 3);
    out[((size_t)n * TDEC + t) * VOC + v] = acc[j];
  }
}

// ---------------- final h_n / c_n ----------------
__global__ void __launch_bounds__(256)
k_final(const float* h0, const float* c0, const float* h1, const float* c1, float* out) {
  int e = blockIdx.x * blockDim.x + threadIdx.x;
  if (e < NB * HID) {
    size_t base = (size_t)NB * TDEC * VOC;
    out[base + e] = h0[e];
    out[base + NB * HID + e] = h1[e];
    out[base + 2 * NB * HID + e] = c0[e];
    out[base + 3 * NB * HID + e] = c1[e];
  }
}

extern "C" void kernel_launch(void* const* d_in, const int* in_sizes, int n_in,
                              void* d_out, int out_size, void* d_ws, size_t ws_size,
                              hipStream_t stream) {
  (void)in_sizes; (void)n_in; (void)out_size; (void)ws_size;
  const int*   src_len = (const int*)d_in[0];
  const float* enc     = (const float*)d_in[1];
  const float* h_0     = (const float*)d_in[2];
  const float* c_0     = (const float*)d_in[3];
  const int*   target  = (const int*)d_in[4];
  const float* emb     = (const float*)d_in[5];
  const float* Wih0    = (const float*)d_in[6];
  const float* Whh0    = (const float*)d_in[7];
  const float* bih0    = (const float*)d_in[8];
  const float* bhh0    = (const float*)d_in[9];
  const float* Wih1    = (const float*)d_in[10];
  const float* Whh1    = (const float*)d_in[11];
  const float* bih1    = (const float*)d_in[12];
  const float* bhh1    = (const float*)d_in[13];
  const float* Wa      = (const float*)d_in[14];
  const float* Wc      = (const float*)d_in[15];
  const float* Wo      = (const float*)d_in[16];
  float* out = (float*)d_out;

  char* p = (char*)d_ws;
  auto take = [&](size_t bytes) { char* r = p; p += (bytes + 255) & ~(size_t)255; return r; };
  unsigned short* W0p  = (unsigned short*)take((size_t)3072 * 4096 * 2);
  unsigned short* W1p  = (unsigned short*)take((size_t)2048 * 4096 * 2);
  unsigned short* Wap  = (unsigned short*)take((size_t)1024 * 1024 * 2);
  unsigned short* Wcp  = (unsigned short*)take((size_t)2048 * 1024 * 2);
  unsigned short* Wop  = (unsigned short*)take((size_t)1024 * 32000 * 2);
  unsigned short* XF   = (unsigned short*)take((size_t)NB * 3072 * 2);
  unsigned short* H0F  = (unsigned short*)take((size_t)NB * HID * 2);
  unsigned short* H1Fa = (unsigned short*)take((size_t)NB * HID * 2);
  unsigned short* H1Fb = (unsigned short*)take((size_t)NB * HID * 2);
  unsigned short* QF   = (unsigned short*)take((size_t)NB * HID * 2);
  unsigned short* CF   = (unsigned short*)take((size_t)NB * 2048 * 2);
  unsigned short* AF   = (unsigned short*)take((size_t)NB * HID * 2);
  float* h0s  = (float*)take((size_t)NB * HID * 4);
  float* c0s  = (float*)take((size_t)NB * HID * 4);
  float* h1s  = (float*)take((size_t)NB * HID * 4);
  float* c1s  = (float*)take((size_t)NB * HID * 4);
  float* qbuf = (float*)take((size_t)NB * HID * 4);
  float* ctxb = (float*)take((size_t)NB * HID * 4);

  k_pack<<<8192, 256, 0, stream>>>(Wih0, Whh0, Wih1, Whh1, Wa, Wc, Wo,
                                   W0p, W1p, Wap, Wcp, Wop);
  k_init<<<256, 256, 0, stream>>>(target, emb, h_0, c_0, XF, H1Fa, h0s, c0s, h1s, c1s);

  for (int t = 0; t < TDEC; ++t) {
    unsigned short* H1cur = (t & 1) ? H1Fb : H1Fa;
    unsigned short* H1nxt = (t & 1) ? H1Fa : H1Fb;
    k_lstm0<<<8, 256, 0, stream>>>(XF, W0p, bih0, bhh0, h0s, c0s, H0F);
    k_lstm1<<<8, 256, 0, stream>>>(H0F, H1cur, W1p, bih1, bhh1, h1s, c1s, H1nxt);
    k_attn<<<1, 512, 0, stream>>>(Wap, Wcp, enc, src_len, target, emb,
                                  h0s, h1s, qbuf, ctxb, QF, CF, AF, XF, t);
    k_logits<<<250, 256, 0, stream>>>(AF, Wop, out, t);
  }
  k_final<<<64, 256, 0, stream>>>(h0s, c0s, h1s, c1s, out);
}
